// BertSelfAttention_14869176778771
// MI455X (gfx1250) — matrix operand order repaired and checked by reading, not compiled, not run on hardware
//
#include <hip/hip_runtime.h>
#include <hip/hip_bf16.h>
#include <stdint.h>

// ---- problem dims (fixed by the reference) ----
#define NB   2
#define NS   2048
#define NH   16
#define DHD  64
#define HID  1024

typedef __attribute__((ext_vector_type(16))) __bf16 v16bf;
typedef __attribute__((ext_vector_type(8)))  float  v8f;

static __device__ __forceinline__ __bf16 f2bf(float f) {
  union { float f; uint32_t u; } a; a.f = f;
  uint32_t r = a.u + 0x7fffu + ((a.u >> 16) & 1u);   // round-to-nearest-even
  union { uint16_t u; __bf16 b; } c; c.u = (uint16_t)(r >> 16);
  return c.b;
}

static __device__ __forceinline__ v8f wmma_bf16(v16bf a, v16bf b, v8f c) {
  return __builtin_amdgcn_wmma_f32_16x16x32_bf16(false, a, false, b, (short)0, c,
                                                 false, false);
}

// A fragment: 16x32 (MxK) bf16 from a row-major [16][>=32] tile, row stride ld.
// ISA layout: lanes 0-15 row M=lane, K 0..7 then 16..23; lanes 16-31 K 8..15, 24..31.
static __device__ __forceinline__ v16bf load_fragA(const __bf16* tile, int ld) {
  const int lane = threadIdx.x & 31;
  const __bf16* p = tile + (size_t)(lane & 15) * ld + ((lane < 16) ? 0 : 8);
  union { v16bf v; uint4 q[2]; } u;
  u.q[0] = *(const uint4*)(p);
  u.q[1] = *(const uint4*)(p + 16);
  return u.v;
}

// B fragment: 32x16 (KxN) bf16 where memory holds the transpose row-major:
// tile[n][k], row stride ld.  ISA layout: lanes 0-15 col N=lane, K 0..7 then 16..23;
// lanes 16-31 K 8..15, 24..31.
static __device__ __forceinline__ v16bf load_fragB(const __bf16* tile, int ld) {
  const int lane = threadIdx.x & 31;
  const __bf16* p = tile + (size_t)(lane & 15) * ld + ((lane < 16) ? 0 : 8);
  union { v16bf v; uint4 q[2]; } u;
  u.q[0] = *(const uint4*)(p);
  u.q[1] = *(const uint4*)(p + 16);
  return u.v;
}

// all-ones bf16 B fragment: P @ ones gives per-row sums in every C column
static __device__ __forceinline__ v16bf ones_frag() {
  union { v16bf v; uint32_t w[8]; } u;
#pragma unroll
  for (int i = 0; i < 8; ++i) u.w[i] = 0x3f803f80u;   // bf16(1.0) x2
  return u.v;
}

static __device__ __forceinline__ float red_max16(float x) {
  x = fmaxf(x, __shfl_xor(x, 1, 32));
  x = fmaxf(x, __shfl_xor(x, 2, 32));
  x = fmaxf(x, __shfl_xor(x, 4, 32));
  x = fmaxf(x, __shfl_xor(x, 8, 32));
  return x;
}

// CDNA5 async copy: global -> LDS, 16B per enabled lane (tracked by ASYNCcnt).
// lds_off is the byte offset inside the kernel's (single) static LDS block,
// which the backend places at group-segment offset 0.
static __device__ __forceinline__ void async_cp16(uint32_t lds_off,
                                                  const __bf16* gptr) {
  asm volatile("global_load_async_to_lds_b128 %0, %1, off"
               :: "v"(lds_off), "v"((uint64_t)(uintptr_t)gptr)
               : "memory");
}

// ---------------- stage 1: fp32 -> bf16 ----------------
__global__ void cvt_f32_bf16(const float* __restrict__ src,
                             __bf16* __restrict__ dst, int n) {
  int i = blockIdx.x * blockDim.x + threadIdx.x;
  int stride = gridDim.x * blockDim.x;
  for (; i < n; i += stride) dst[i] = f2bf(src[i]);
}

// ---------------- stage 2: QKV projection GEMM ----------------
// One wave: 16 rows x 64 cols (one head) of   out = X @ W^T + bias.
// W stored [o][i] row-major => exactly the B-fragment memory layout.
// Q,K written [b,h,s,d] bf16; V written transposed [b,h,d,s] bf16.
__global__ __launch_bounds__(128)
void qkv_gemm(const __bf16* __restrict__ hb, const __bf16* __restrict__ kbin,
              const __bf16* __restrict__ vbin,
              const __bf16* __restrict__ Wqb, const __bf16* __restrict__ Wkb,
              const __bf16* __restrict__ Wvb,
              const float* __restrict__ bq, const float* __restrict__ bk,
              const float* __restrict__ bv,
              __bf16* __restrict__ qh, __bf16* __restrict__ kh,
              __bf16* __restrict__ vt)
{
  const int lane = threadIdx.x & 31;
  const int wave = threadIdx.x >> 5;
  const int mat  = blockIdx.z >> 1;          // 0=Q 1=K 2=V
  const int b    = blockIdx.z & 1;
  const int h    = blockIdx.y;
  const int n0   = (blockIdx.x * 4 + wave) * 16;

  const __bf16* src  = (mat == 0) ? hb : (mat == 1) ? kbin : vbin;
  const __bf16* W    = (mat == 0) ? Wqb : (mat == 1) ? Wkb : Wvb;
  const float*  bias = (mat == 0) ? bq : (mat == 1) ? bk : bv;

  const __bf16* A  = src + ((size_t)b * NS + n0) * HID;
  const __bf16* Wh = W + (size_t)h * DHD * HID;

  v8f z = {};
  v8f acc[4] = {z, z, z, z};

  for (int kk = 0; kk < HID; kk += 32) {
    v16bf a = load_fragA(A + kk, HID);
#pragma unroll
    for (int f = 0; f < 4; ++f) {
      v16bf bf = load_fragB(Wh + (size_t)(16 * f) * HID + kk, HID);
      acc[f] = wmma_bf16(a, bf, acc[f]);
    }
  }

  const int rbase = (lane < 16) ? 0 : 8;
  const int col   = lane & 15;
  if (mat < 2) {
    __bf16* dst = ((mat == 0) ? qh : kh) + ((size_t)(b * NH + h) * NS + n0) * DHD;
#pragma unroll
    for (int f = 0; f < 4; ++f) {
      float bb = bias[h * DHD + 16 * f + col];
#pragma unroll
      for (int r = 0; r < 8; ++r)
        dst[(size_t)(r + rbase) * DHD + 16 * f + col] = f2bf(acc[f][r] + bb);
    }
  } else {
    __bf16* dst = vt + (size_t)(b * NH + h) * DHD * NS;
#pragma unroll
    for (int f = 0; f < 4; ++f) {
      float bb = bias[h * DHD + 16 * f + col];
#pragma unroll
      for (int r = 0; r < 8; ++r)
        dst[(size_t)(16 * f + col) * NS + (n0 + r + rbase)] = f2bf(acc[f][r] + bb);
    }
  }
}

// ---------------- stage 3: flash attention ----------------
// Block = 4 waves sharing one (b,h); each wave owns a 16-row Q tile.
// K/V chunks (32 columns) are staged into LDS once per block via the CDNA5
// async-to-LDS path, double-buffered; WMMA operands then come from LDS.
//
// LDS element layout (single static block, base offset 0):
//   [0,        4096)  : K stage, 2 bufs x [32 tok][64 d]   (2 x 4KB)
//   [4096,     8192)  : V stage, 2 bufs x [64 d][32 tok]   (2 x 4KB)
//   [8192,    10240)  : P tiles, 4 waves x [16][32]        (4 x 1KB)
#define LDS_K(buf)   ((buf) * 2048)
#define LDS_V(buf)   (4096 + (buf) * 2048)
#define LDS_P(w)     (8192 + (w) * 512)

__global__ __launch_bounds__(128)
void flash_attn(const __bf16* __restrict__ qh, const __bf16* __restrict__ kh,
                const __bf16* __restrict__ vt, const float* __restrict__ spatial,
                float* __restrict__ out)
{
  __shared__ __bf16 smem[2 * 2048 + 2 * 2048 + 4 * 512];   // 20 KB
  const int tid  = threadIdx.x;
  const int lane = tid & 31;
  const int wave = tid >> 5;
  const int b  = blockIdx.z;
  const int h  = blockIdx.y;
  const int m0 = (blockIdx.x * 4 + wave) * 16;

  const __bf16* qtp = qh + ((size_t)(b * NH + h) * NS + m0) * DHD;
  const __bf16* khp = kh + (size_t)(b * NH + h) * NS * DHD;
  const __bf16* vtp = vt + (size_t)(b * NH + h) * DHD * NS;
  const float*  spp = spatial + ((size_t)(b * NH + h) * NS + m0) * NS;
  float* outp = out + (size_t)b * NS * HID;

  const int col   = lane & 15;
  const int rbase = (lane < 16) ? 0 : 8;
  __bf16* lp = smem + LDS_P(wave);

  v16bf qa0 = load_fragA(qtp, DHD);         // K = d 0..31
  v16bf qa1 = load_fragA(qtp + 32, DHD);    // K = d 32..63
  const v16bf onesB = ones_frag();

  v8f z = {};
  v8f Oacc[4] = {z, z, z, z};
  v8f Lacc = z;                              // row-sum accumulator via WMMA
  float mrow[8];
#pragma unroll
  for (int r = 0; r < 8; ++r) mrow[r] = -3.0e38f;

  // per-thread async staging of one 32-column K/V chunk (4 x 16B per thread)
  auto issue_stage = [&](int buf, int c0) {
#pragma unroll
    for (int j = 0; j < 2; ++j) {
      int i = tid * 2 + j;                                   // 256 x 16B = 4KB
      // K chunk: rows c0..c0+31, all 64 d -> contiguous in [tok][d] storage
      async_cp16((uint32_t)(LDS_K(buf) + i * 8) * 2, khp + (size_t)c0 * DHD + i * 8);
      // V chunk: vt[d][tok], 64 rows x 32 cols, row stride NS
      int d = i >> 2, q = i & 3;
      async_cp16((uint32_t)(LDS_V(buf) + d * 32 + q * 8) * 2,
                 vtp + (size_t)d * NS + c0 + q * 8);
    }
  };

  issue_stage(0, 0);

  for (int c0 = 0; c0 < NS; c0 += 32) {
    const int buf = (c0 >> 5) & 1;
    if (c0 + 32 < NS) {
      issue_stage(buf ^ 1, c0 + 32);
      asm volatile("s_wait_asynccnt 0x4" ::: "memory");  // current buf landed
    } else {
      asm volatile("s_wait_asynccnt 0x0" ::: "memory");
    }
    __syncthreads();   // all waves' staging of `buf` visible

    const __bf16* ksm = smem + LDS_K(buf);   // [32 tok][64 d]
    const __bf16* vsm = smem + LDS_V(buf);   // [64 d][32 tok]

    // scores: S[16 x 32] = Q(16x64) @ K^T(64x32), two 16-col subtiles
    v16bf k00 = load_fragB(ksm, DHD);
    v16bf k01 = load_fragB(ksm + 32, DHD);
    v16bf k10 = load_fragB(ksm + 16 * DHD, DHD);
    v16bf k11 = load_fragB(ksm + 16 * DHD + 32, DHD);
    v8f s0 = wmma_bf16(qa0, k00, z); s0 = wmma_bf16(qa1, k01, s0);
    v8f s1 = wmma_bf16(qa0, k10, z); s1 = wmma_bf16(qa1, k11, s1);

    // scale + spatial bias + online softmax (row r lives in one VGPR / 16 lanes)
#pragma unroll
    for (int r = 0; r < 8; ++r) {
      const float* sr = spp + (size_t)(r + rbase) * NS + c0;
      float a0 = s0[r] * 0.125f + sr[col];
      float a1 = s1[r] * 0.125f + sr[16 + col];
      float mnew = fmaxf(mrow[r], red_max16(fmaxf(a0, a1)));
      float alpha = __expf(mrow[r] - mnew);
      mrow[r] = mnew;
      float p0 = __expf(a0 - mnew);
      float p1 = __expf(a1 - mnew);
      Oacc[0][r] *= alpha; Oacc[1][r] *= alpha;
      Oacc[2][r] *= alpha; Oacc[3][r] *= alpha;
      Lacc[r] *= alpha;
      lp[(r + rbase) * 32 + col]      = f2bf(p0);   // C-layout -> row-major LDS
      lp[(r + rbase) * 32 + 16 + col] = f2bf(p1);
    }
    asm volatile("s_wait_dscnt 0x0" ::: "memory");   // wave-private LDS RAW

    v16bf pa = load_fragA(lp, 32);                   // P as 16x32 A fragment
    Lacc = wmma_bf16(pa, onesB, Lacc);               // row sums via WMMA
#pragma unroll
    for (int f = 0; f < 4; ++f) {
      v16bf vb = load_fragB(vsm + (size_t)(16 * f) * 32, 32);
      Oacc[f] = wmma_bf16(pa, vb, Oacc[f]);
    }
  }

  // epilogue: normalize and scatter to [B,S,H*DH] fp32
#pragma unroll
  for (int r = 0; r < 8; ++r) {
    float inv = 1.0f / Lacc[r];
    float* po = outp + (size_t)(m0 + r + rbase) * HID + h * DHD;
    po[col]      = Oacc[0][r] * inv;
    po[16 + col] = Oacc[1][r] * inv;
    po[32 + col] = Oacc[2][r] * inv;
    po[48 + col] = Oacc[3][r] * inv;
  }
}

// ---------------- host launcher ----------------
extern "C" void kernel_launch(void* const* d_in, const int* in_sizes, int n_in,
                              void* d_out, int out_size, void* d_ws, size_t ws_size,
                              hipStream_t stream) {
  const float* hs = (const float*)d_in[0];
  const float* ks = (const float*)d_in[1];
  const float* vs = (const float*)d_in[2];
  const float* sp = (const float*)d_in[3];
  const float* Wq = (const float*)d_in[4];
  const float* bq = (const float*)d_in[5];
  const float* Wk = (const float*)d_in[6];
  const float* bk = (const float*)d_in[7];
  const float* Wv = (const float*)d_in[8];
  const float* bv = (const float*)d_in[9];
  float* out = (float*)d_out;

  const size_t szHKV  = (size_t)NB * NS * HID * sizeof(__bf16);      // 8 MB
  const size_t szW    = (size_t)HID * HID * sizeof(__bf16);          // 2 MB
  const size_t szHead = (size_t)NB * NH * NS * DHD * sizeof(__bf16); // 8 MB

  char* w = (char*)d_ws;
  size_t off = 0;
  __bf16* hb   = (__bf16*)(w + off); off += szHKV;
  __bf16* kbin = (__bf16*)(w + off); off += szHKV;
  __bf16* vbin = (__bf16*)(w + off); off += szHKV;
  __bf16* Wqb  = (__bf16*)(w + off); off += szW;
  __bf16* Wkb  = (__bf16*)(w + off); off += szW;
  __bf16* Wvb  = (__bf16*)(w + off); off += szW;
  __bf16* qh   = (__bf16*)(w + off); off += szHead;
  __bf16* kh   = (__bf16*)(w + off); off += szHead;
  __bf16* vt   = (__bf16*)(w + off); off += szHead;   // V transposed [b,h,d,s]

  const int nAct = NB * NS * HID;
  const int nW   = HID * HID;
  cvt_f32_bf16<<<dim3((nAct + 255) / 256), 256, 0, stream>>>(hs, hb, nAct);
  cvt_f32_bf16<<<dim3((nAct + 255) / 256), 256, 0, stream>>>(ks, kbin, nAct);
  cvt_f32_bf16<<<dim3((nAct + 255) / 256), 256, 0, stream>>>(vs, vbin, nAct);
  cvt_f32_bf16<<<dim3((nW + 255) / 256), 256, 0, stream>>>(Wq, Wqb, nW);
  cvt_f32_bf16<<<dim3((nW + 255) / 256), 256, 0, stream>>>(Wk, Wkb, nW);
  cvt_f32_bf16<<<dim3((nW + 255) / 256), 256, 0, stream>>>(Wv, Wvb, nW);

  // 4 waves/block, 16 rows/wave: x covers S/64 row-tiles, y = head, z = mat*2+b
  qkv_gemm<<<dim3(NS / 64, NH, 6), 128, 0, stream>>>(
      hb, kbin, vbin, Wqb, Wkb, Wvb, bq, bk, bv, qh, kh, vt);

  flash_attn<<<dim3(NS / 64, NH, NB), 128, 0, stream>>>(qh, kh, vt, sp, out);
}
